// LeGNN4_61598420959267
// MI455X (gfx1250) — matrix operand order repaired and checked by reading, not compiled, not run on hardware
//
#include <hip/hip_runtime.h>
#include <hip/hip_bf16.h>

typedef __bf16 bf16;
typedef __attribute__((ext_vector_type(16))) __bf16 v16bf;
typedef __attribute__((ext_vector_type(8))) float v8f;

// ---------------------------------------------------------------------------
// helpers
// ---------------------------------------------------------------------------
__device__ __forceinline__ v16bf make_v16bf(uint4 lo, uint4 hi) {
  union { unsigned u[8]; v16bf v; } t;
  t.u[0] = lo.x; t.u[1] = lo.y; t.u[2] = lo.z; t.u[3] = lo.w;
  t.u[4] = hi.x; t.u[5] = hi.y; t.u[6] = hi.z; t.u[7] = hi.w;
  return t.v;
}

__device__ __forceinline__ v8f wmma_bf16(v16bf a, v16bf b, v8f c) {
  // D(f32 16x16) = A(bf16 16x32) * B(bf16 32x16) + C
  return __builtin_amdgcn_wmma_f32_16x16x32_bf16(false, a, false, b, (short)0, c,
                                                 false, false);
}

// async copy of 16 contiguous bytes global -> LDS (ASYNCcnt-tracked DMA)
__device__ __forceinline__ void async_copy_b128(unsigned lds_off,
                                                const void* gptr) {
  asm volatile("global_load_async_to_lds_b128 %0, %1, off" ::"v"(lds_off),
               "v"(gptr)
               : "memory");
}
__device__ __forceinline__ void wait_async_le1() {
  asm volatile("s_wait_asynccnt 0x1" ::: "memory");
}
__device__ __forceinline__ void wait_async_0() {
  asm volatile("s_wait_asynccnt 0x0" ::: "memory");
}
// flat shared address -> raw LDS byte offset (aperture keeps offset in [31:0])
__device__ __forceinline__ unsigned lds_off_of(const void* p) {
  return (unsigned)(size_t)p;
}

// order-preserving float<->uint map for atomicMax-based segment max
__device__ __forceinline__ unsigned encf(float f) {
  unsigned u = __float_as_uint(f);
  return (u & 0x80000000u) ? ~u : (u | 0x80000000u);
}
__device__ __forceinline__ float decf(unsigned e) {
  return (e & 0x80000000u) ? __uint_as_float(e & 0x7FFFFFFFu)
                           : __uint_as_float(~e);
}

// ---------------------------------------------------------------------------
// elementwise utility kernels
// ---------------------------------------------------------------------------
__global__ void f2bf_kernel(const float* __restrict__ s, bf16* __restrict__ d,
                            size_t n) {
  size_t i = (size_t)blockIdx.x * 256 + threadIdx.x;
  if (i < n) d[i] = (bf16)s[i];
}

__global__ void fill_u32_kernel(unsigned* __restrict__ p, unsigned v, size_t n) {
  size_t i = (size_t)blockIdx.x * 256 + threadIdx.x;
  if (i < n) p[i] = v;
}

// pack f32 [nmat][K][C] -> bf16 [nmat][K/32][C][32] (k-innermost, WMMA-B layout)
__global__ void pack_kernel(const float* __restrict__ W, bf16* __restrict__ out,
                            int K, int C, size_t total) {
  size_t i = (size_t)blockIdx.x * 256 + threadIdx.x;
  if (i >= total) return;
  size_t per = (size_t)K * C;
  size_t mat = i / per;
  size_t r = i % per;
  int k0 = (int)(r / ((size_t)C * 32));
  int rem = (int)(r % ((size_t)C * 32));
  int c = rem / 32, k = rem & 31;
  out[i] = (bf16)W[mat * per + (size_t)(k0 * 32 + k) * C + c];
}

// gact = bf16(gelu(aggsum / (s + 1e-16)))
__global__ void gelu_kernel(const float* __restrict__ agg,
                            const float* __restrict__ sbuf,
                            bf16* __restrict__ gact, size_t n) {
  size_t i = (size_t)blockIdx.x * 256 + threadIdx.x;
  if (i >= n) return;
  size_t node = i >> 8;
  int h = (int)((i >> 5) & 7);
  float x = agg[i] / (sbuf[node * 8 + h] + 1e-16f);
  float g = 0.5f * x *
            (1.0f + tanhf(0.7978845608028654f * (x + 0.044715f * x * x * x)));
  gact[i] = (bf16)g;
}

// ---------------------------------------------------------------------------
// Generic bf16 WMMA GEMM: C[M,256] = A[M,K] * Bp[K,256] + bias
// Bp is pre-packed [K/32][256][32] so each 32(K)x64(N) slab is one contiguous
// 4KB block: staged by double-buffered async global->LDS DMA overlapped with
// the WMMA math of the previous slab.
// block = 256 thr = 8 waves; block tile 128(M) x 64(N); wave tile 16 x 64.
// ---------------------------------------------------------------------------
__global__ __launch_bounds__(256) void gemm_kernel(
    const bf16* __restrict__ A, const bf16* __restrict__ Bp,
    const float* __restrict__ bias, float* __restrict__ outF,
    bf16* __restrict__ outB, int M, int K) {
  __shared__ bf16 bsh[2][2048];  // double-buffered 64(N) x 32(K) slabs
  int tid = threadIdx.x;
  int wave = tid >> 5, lane = tid & 31;
  int lh = lane & 15, grp = lane >> 4;
  int colBase = blockIdx.x * 64;
  int rowBase = blockIdx.y * 128 + wave * 16;

  v8f acc[4] = {};

  int arow = rowBase + lh;
  if (arow >= M) arow = M - 1;
  const bf16* aptr = A + (size_t)arow * K + (grp ? 16 : 0);

  unsigned ldsBase = lds_off_of(&bsh[0][0]);
  auto stage = [&](int s, int buf) {
    const bf16* gsrc = Bp + ((size_t)s * 256 + colBase) * 32 + (size_t)tid * 8;
    async_copy_b128(ldsBase + (unsigned)buf * 4096u + (unsigned)tid * 16u,
                    gsrc);
  };

  int nsteps = K >> 5;
  stage(0, 0);
  for (int s = 0; s < nsteps; s++) {
    int cur = s & 1;
    if (s + 1 < nsteps) {
      stage(s + 1, cur ^ 1);  // DMA next slab while computing this one
      wait_async_le1();
    } else {
      wait_async_0();
    }
    __syncthreads();

    int k0 = s << 5;
    // A fragment (16x32 bf16): lanes<16 K 0..15, lanes>=16 K 16..31 (matches B)
    uint4 alo = *reinterpret_cast<const uint4*>(aptr + k0);
    uint4 ahi = *reinterpret_cast<const uint4*>(aptr + k0 + 8);
    v16bf afrag = make_v16bf(alo, ahi);
    __builtin_prefetch(aptr + k0 + 64, 0, 3);

    v16bf bfrag[4];
#pragma unroll
    for (int nt = 0; nt < 4; nt++) {
      const uint4* bp = reinterpret_cast<const uint4*>(
          &bsh[cur][(nt * 16 + lh) * 32 + grp * 16]);
      bfrag[nt] = make_v16bf(bp[0], bp[1]);
    }
#pragma unroll
    for (int nt = 0; nt < 4; nt++) acc[nt] = wmma_bf16(afrag, bfrag[nt], acc[nt]);
    __syncthreads();
  }

#pragma unroll
  for (int nt = 0; nt < 4; nt++) {
#pragma unroll
    for (int r = 0; r < 8; r++) {
      int row = rowBase + r + grp * 8;
      int col = colBase + nt * 16 + lh;
      if (row < M) {
        float v = acc[nt][r] + bias[col];
        if (outF) outF[(size_t)row * 256 + col] = v;
        if (outB) outB[(size_t)row * 256 + col] = (bf16)v;
      }
    }
  }
}

// ---------------------------------------------------------------------------
// Edge pass A: ke = gather(K)[src] @ a_rel + ea ; logit = (q . ke) * p/sqrt(DK)
// a_rel arrives pre-packed [h][m][k]; staged with one contiguous 16KB async
// DMA. Each wave: one 16-edge tile, 8 heads, 2 WMMAs / head.
// ---------------------------------------------------------------------------
__global__ __launch_bounds__(256) void edge_pass_a(
    const bf16* __restrict__ Kbf, const float* __restrict__ Qf,
    const float* __restrict__ ea, const int* __restrict__ src,
    const int* __restrict__ dst, const bf16* __restrict__ arelP,
    const float* __restrict__ prel, float* __restrict__ logit,
    unsigned* __restrict__ mbuf, int offS, int offD, int Ee) {
  __shared__ bf16 alds[8192];  // [h][m(out)][k]
  int tid = threadIdx.x;
  {
    unsigned ldst = lds_off_of(&alds[0]) + (unsigned)tid * 64u;
    const bf16* gsrc = arelP + (size_t)tid * 32;
#pragma unroll
    for (int j = 0; j < 4; j++)
      async_copy_b128(ldst + (unsigned)j * 16u, gsrc + j * 8);
    wait_async_0();
  }
  __syncthreads();

  int wave = tid >> 5, lane = tid & 31;
  int lh = lane & 15, grp = lane >> 4;
  int e0 = (blockIdx.x * 8 + wave) * 16;
  if (e0 >= Ee) return;

  int eA = e0 + lh;
  if (eA >= Ee) eA = Ee - 1;
  size_t srow = (size_t)(offS + src[eA]) * 256;
  const bf16* kp = Kbf + srow + (grp ? 16 : 0);

  for (int h = 0; h < 8; h++) {
    uint4 alo = *reinterpret_cast<const uint4*>(kp + h * 32);
    uint4 ahi = *reinterpret_cast<const uint4*>(kp + h * 32 + 8);
    v16bf A = make_v16bf(alo, ahi);
    const uint4* b0 =
        reinterpret_cast<const uint4*>(&alds[(h * 32 + lh) * 32 + grp * 16]);
    const uint4* b1 = reinterpret_cast<const uint4*>(
        &alds[(h * 32 + lh + 16) * 32 + grp * 16]);
    v8f z0 = {}, z1 = {};
    v8f c0 = wmma_bf16(A, make_v16bf(b0[0], b0[1]), z0);
    v8f c1 = wmma_bf16(A, make_v16bf(b1[0], b1[1]), z1);
    float sc = prel[h] * 0.17677669529663687f;  // 1/sqrt(32)
#pragma unroll
    for (int r = 0; r < 8; r++) {
      int er = e0 + r + grp * 8;
      bool valid = er < Ee;
      int eC = valid ? er : Ee - 1;
      size_t drow = (size_t)(offD + dst[eC]) * 256;
      size_t eb = (size_t)eC * 256 + h * 32 + lh;
      float p = (c0[r] + ea[eb]) * Qf[drow + h * 32 + lh] +
                (c1[r] + ea[eb + 16]) * Qf[drow + h * 32 + 16 + lh];
      p += __shfl_xor(p, 1);
      p += __shfl_xor(p, 2);
      p += __shfl_xor(p, 4);
      p += __shfl_xor(p, 8);
      if (lh == 0 && valid) {
        float lg = p * sc;
        logit[(size_t)er * 8 + h] = lg;
        atomicMax(&mbuf[(drow >> 5) + h], encf(lg));
      }
    }
  }
}

// ---------------------------------------------------------------------------
// Edge pass B: ve = gather(V)[src] @ m_rel + ea ; ex = exp(logit - max);
// atomic accumulate ex and ex*ve per destination node.
// ---------------------------------------------------------------------------
__global__ __launch_bounds__(256) void edge_pass_b(
    const bf16* __restrict__ Vbf, const float* __restrict__ ea,
    const int* __restrict__ src, const int* __restrict__ dst,
    const bf16* __restrict__ mrelP, const float* __restrict__ logit,
    const unsigned* __restrict__ mbuf, float* __restrict__ sbuf,
    float* __restrict__ agg, int offS, int offD, int Ee) {
  __shared__ bf16 alds[8192];
  int tid = threadIdx.x;
  {
    unsigned ldst = lds_off_of(&alds[0]) + (unsigned)tid * 64u;
    const bf16* gsrc = mrelP + (size_t)tid * 32;
#pragma unroll
    for (int j = 0; j < 4; j++)
      async_copy_b128(ldst + (unsigned)j * 16u, gsrc + j * 8);
    wait_async_0();
  }
  __syncthreads();

  int wave = tid >> 5, lane = tid & 31;
  int lh = lane & 15, grp = lane >> 4;
  int e0 = (blockIdx.x * 8 + wave) * 16;
  if (e0 >= Ee) return;

  int eA = e0 + lh;
  if (eA >= Ee) eA = Ee - 1;
  size_t srow = (size_t)(offS + src[eA]) * 256;
  const bf16* vp = Vbf + srow + (grp ? 16 : 0);

  for (int h = 0; h < 8; h++) {
    uint4 alo = *reinterpret_cast<const uint4*>(vp + h * 32);
    uint4 ahi = *reinterpret_cast<const uint4*>(vp + h * 32 + 8);
    v16bf A = make_v16bf(alo, ahi);
    const uint4* b0 =
        reinterpret_cast<const uint4*>(&alds[(h * 32 + lh) * 32 + grp * 16]);
    const uint4* b1 = reinterpret_cast<const uint4*>(
        &alds[(h * 32 + lh + 16) * 32 + grp * 16]);
    v8f z0 = {}, z1 = {};
    v8f c0 = wmma_bf16(A, make_v16bf(b0[0], b0[1]), z0);
    v8f c1 = wmma_bf16(A, make_v16bf(b1[0], b1[1]), z1);
#pragma unroll
    for (int r = 0; r < 8; r++) {
      int er = e0 + r + grp * 8;
      bool valid = er < Ee;
      int eC = valid ? er : Ee - 1;
      size_t drow = (size_t)(offD + dst[eC]) * 256;
      float lg = logit[(size_t)eC * 8 + h];
      float mm = decf(mbuf[(drow >> 5) + h]);
      float ex = __expf(lg - mm);
      if (valid) {
        size_t eb = (size_t)eC * 256 + h * 32 + lh;
        if (lh == 0) atomicAdd(&sbuf[(drow >> 5) + h], ex);
        atomicAdd(&agg[drow + h * 32 + lh], ex * (c0[r] + ea[eb]));
        atomicAdd(&agg[drow + h * 32 + 16 + lh], ex * (c1[r] + ea[eb + 16]));
      }
    }
  }
}

// ---------------------------------------------------------------------------
// LayerNorm + sigmoid skip gate: one wave per row (8 rows / block).
// writes f32 output and bf16 copy (next layer's GEMM A).
// ---------------------------------------------------------------------------
__global__ __launch_bounds__(256) void ln_kernel(
    const float* __restrict__ o, const float* __restrict__ hprev,
    const float* __restrict__ skipg, const float* __restrict__ lng,
    const float* __restrict__ lnb, float* __restrict__ outF,
    bf16* __restrict__ outB, int M) {
  int wave = threadIdx.x >> 5, lane = threadIdx.x & 31;
  int row = blockIdx.x * 8 + wave;
  if (row >= M) return;
  float g = 1.0f / (1.0f + __expf(-skipg[0]));
  const float* op = o + (size_t)row * 256;
  const float* hp = hprev + (size_t)row * 256;
  float x[8], s = 0.f, s2 = 0.f;
#pragma unroll
  for (int j = 0; j < 8; j++) {
    int d = j * 32 + lane;
    float v = g * op[d] + (1.0f - g) * hp[d];
    x[j] = v;
    s += v;
    s2 += v * v;
  }
  for (int off = 16; off; off >>= 1) {
    s += __shfl_xor(s, off);
    s2 += __shfl_xor(s2, off);
  }
  float mu = s * (1.0f / 256.0f);
  float var = s2 * (1.0f / 256.0f) - mu * mu;
  float inv = rsqrtf(var + 1e-5f);
#pragma unroll
  for (int j = 0; j < 8; j++) {
    int d = j * 32 + lane;
    float y = (x[j] - mu) * inv * lng[d] + lnb[d];
    outF[(size_t)row * 256 + d] = y;
    outB[(size_t)row * 256 + d] = (bf16)y;
  }
}

// ---------------------------------------------------------------------------
// host orchestration
// ---------------------------------------------------------------------------
extern "C" void kernel_launch(void* const* d_in, const int* in_sizes, int n_in,
                              void* d_out, int out_size, void* d_ws,
                              size_t ws_size, hipStream_t stream) {
  (void)in_sizes; (void)n_in; (void)out_size; (void)ws_size;

  static const int Ns[4] = {20000, 60000, 8000, 400};
  static const int offN[4] = {0, 20000, 80000, 88000};
  const int NT = 88400;
  static const int Es[4] = {60000, 250000, 120000, 16000};
  static const int offE[4] = {0, 60000, 310000, 430000};
  const size_t ETOT = 446000;
  static const int esrc[4] = {1, 2, 1, 2};
  static const int edst[4] = {0, 1, 3, 3};

  const int* ei[4];
  const float* et[4];
  for (int e = 0; e < 4; e++) {
    ei[e] = (const int*)d_in[4 + 2 * e];
    et[e] = (const float*)d_in[5 + 2 * e];
  }
  const float* Wkqv = (const float*)d_in[12];
  const float* b_kqv = (const float*)d_in[13];
  const float* a_rel = (const float*)d_in[14];
  const float* m_rel = (const float*)d_in[15];
  const float* p_rel = (const float*)d_in[16];
  const float* Wout = (const float*)d_in[17];
  const float* b_out = (const float*)d_in[18];
  const float* skip_g = (const float*)d_in[19];
  const float* elin_W = (const float*)d_in[20];
  const float* elin_b = (const float*)d_in[21];
  const float* ln_g = (const float*)d_in[22];
  const float* ln_b = (const float*)d_in[23];

  const size_t SZH = (size_t)NT * 256;

  char* pws = (char*)d_ws;
  auto alloc = [&](size_t bytes) -> char* {
    char* r = pws;
    pws += (bytes + 255) & ~(size_t)255;
    return r;
  };
  float* hA = (float*)alloc(SZH * 4);
  float* hB = (float*)alloc(SZH * 4);
  bf16* hbf = (bf16*)alloc(SZH * 2);
  bf16* Kbf = (bf16*)alloc(SZH * 2);
  bf16* Vbf = (bf16*)alloc(SZH * 2);
  float* Qf = (float*)alloc(SZH * 4);
  float* agg = (float*)alloc(SZH * 4);  // also reused as Wout GEMM output
  bf16* gact = (bf16*)alloc(SZH * 2);
  float* sbuf = (float*)alloc((size_t)NT * 8 * 4);
  unsigned* mbuf = (unsigned*)alloc((size_t)NT * 8 * 4);
  float* logit = (float*)alloc(ETOT * 8 * 4);
  float* ea = (float*)alloc(ETOT * 256 * 4);
  bf16* etbf = (bf16*)alloc(ETOT * 64 * 2);
  bf16* WkqvB = (bf16*)alloc((size_t)2 * 4 * 3 * 65536 * 2);  // packed
  bf16* WoutB = (bf16*)alloc((size_t)2 * 4 * 65536 * 2);      // packed
  bf16* elinWB = (bf16*)alloc((size_t)4 * 64 * 256 * 2);      // packed
  bf16* arelB = (bf16*)alloc((size_t)2 * 4 * 8 * 1024 * 2);   // packed [h][m][k]
  bf16* mrelB = (bf16*)alloc((size_t)2 * 4 * 8 * 1024 * 2);   // packed [h][m][k]

  auto cvt = [&](const float* s, bf16* d, size_t n) {
    f2bf_kernel<<<(unsigned)((n + 255) / 256), 256, 0, stream>>>(s, d, n);
  };
  auto pack = [&](const float* s, bf16* d, int K, int C, size_t n) {
    pack_kernel<<<(unsigned)((n + 255) / 256), 256, 0, stream>>>(s, d, K, C, n);
  };
  auto fillu = [&](void* d, unsigned v, size_t n) {
    fill_u32_kernel<<<(unsigned)((n + 255) / 256), 256, 0, stream>>>(
        (unsigned*)d, v, n);
  };
  auto gemm = [&](const bf16* A, const bf16* Bp, const float* bias, float* oF,
                  bf16* oB, int M, int K) {
    dim3 g(4, (unsigned)((M + 127) / 128));
    gemm_kernel<<<g, 256, 0, stream>>>(A, Bp, bias, oF, oB, M, K);
  };

  // gather inputs into contiguous h buffer + bf16 copy
  for (int t = 0; t < 4; t++)
    hipMemcpyAsync(hA + (size_t)offN[t] * 256, d_in[t],
                   (size_t)Ns[t] * 256 * 4, hipMemcpyDeviceToDevice, stream);
  cvt(hA, hbf, SZH);

  // weight packing (bf16 + WMMA-friendly layouts)
  pack(Wkqv, WkqvB, 256, 256, (size_t)2 * 4 * 3 * 65536);
  pack(Wout, WoutB, 256, 256, (size_t)2 * 4 * 65536);
  pack(elin_W, elinWB, 64, 256, (size_t)4 * 64 * 256);
  pack(a_rel, arelB, 32, 32, (size_t)2 * 4 * 8 * 1024);
  pack(m_rel, mrelB, 32, 32, (size_t)2 * 4 * 8 * 1024);
  for (int e = 0; e < 4; e++)
    cvt(et[e], etbf + (size_t)offE[e] * 64, (size_t)Es[e] * 64);

  // ea[e] = et[e] @ elin_W[e] + elin_b[e]   (computed once, reused both layers)
  for (int e = 0; e < 4; e++)
    gemm(etbf + (size_t)offE[e] * 64, elinWB + (size_t)e * 64 * 256,
         elin_b + (size_t)e * 256, ea + (size_t)offE[e] * 256, nullptr, Es[e],
         64);

  for (int l = 0; l < 2; l++) {
    const float* prev = l ? hB : hA;
    float* next = l ? (float*)d_out : hB;

    // K/Q/V projections (K,V -> bf16 for edge WMMA gathers, Q -> f32)
    for (int t = 0; t < 4; t++) {
      for (int c = 0; c < 3; c++) {
        size_t wOff = (((size_t)l * 4 + t) * 3 + c) * 65536;
        const float* bias = b_kqv + (((size_t)l * 4 + t) * 3 + c) * 256;
        const bf16* Ain = hbf + (size_t)offN[t] * 256;
        if (c == 0)
          gemm(Ain, WkqvB + wOff, bias, nullptr, Kbf + (size_t)offN[t] * 256,
               Ns[t], 256);
        else if (c == 1)
          gemm(Ain, WkqvB + wOff, bias, Qf + (size_t)offN[t] * 256, nullptr,
               Ns[t], 256);
        else
          gemm(Ain, WkqvB + wOff, bias, nullptr, Vbf + (size_t)offN[t] * 256,
               Ns[t], 256);
      }
    }

    fillu(agg, 0u, SZH);
    fillu(sbuf, 0u, (size_t)NT * 8);
    fillu(mbuf, 0x007FFFFFu, (size_t)NT * 8);  // enc(-inf)

    // pass A: logits + segment max
    for (int e = 0; e < 4; e++) {
      unsigned blocks = (unsigned)((Es[e] + 127) / 128);
      edge_pass_a<<<blocks, 256, 0, stream>>>(
          Kbf, Qf, ea + (size_t)offE[e] * 256, ei[e], ei[e] + Es[e],
          arelB + ((size_t)l * 4 + e) * 8192, p_rel + ((size_t)l * 4 + e) * 8,
          logit + (size_t)offE[e] * 8, mbuf, offN[esrc[e]], offN[edst[e]],
          Es[e]);
    }
    // pass B: exp + weighted value aggregation
    for (int e = 0; e < 4; e++) {
      unsigned blocks = (unsigned)((Es[e] + 127) / 128);
      edge_pass_b<<<blocks, 256, 0, stream>>>(
          Vbf, ea + (size_t)offE[e] * 256, ei[e], ei[e] + Es[e],
          mrelB + ((size_t)l * 4 + e) * 8192, logit + (size_t)offE[e] * 8,
          mbuf, sbuf, agg, offN[esrc[e]], offN[edst[e]], Es[e]);
    }

    // softmax normalize + gelu -> bf16 activation for Wout GEMM
    gelu_kernel<<<(unsigned)((SZH + 255) / 256), 256, 0, stream>>>(agg, sbuf,
                                                                   gact, SZH);

    // o = gelu(agg) @ Wout + b_out  (into agg buffer, now free)
    for (int t = 0; t < 4; t++)
      gemm(gact + (size_t)offN[t] * 256, WoutB + ((size_t)l * 4 + t) * 65536,
           b_out + ((size_t)l * 4 + t) * 256, agg + (size_t)offN[t] * 256,
           nullptr, Ns[t], 256);

    // skip gate + LayerNorm -> next h (f32) and bf16 copy for next layer
    for (int t = 0; t < 4; t++) {
      unsigned blocks = (unsigned)((Ns[t] + 7) / 8);
      ln_kernel<<<blocks, 256, 0, stream>>>(
          agg + (size_t)offN[t] * 256, prev + (size_t)offN[t] * 256,
          skip_g + (size_t)l * 4 + t, ln_g + (size_t)t * 256,
          ln_b + (size_t)t * 256, next + (size_t)offN[t] * 256,
          hbf + (size_t)offN[t] * 256, Ns[t]);
    }
  }
}
